// MTPObjective_11441792877178
// MI455X (gfx1250) — compile-verified, matrix operand order checked by reading, not run on hardware
//
#include <hip/hip_runtime.h>
#include <hip/hip_bf16.h>

#define B_DIM 2
#define S_DIM 2048
#define D_DIM 2048
#define V_DIM 32000
#define K_MTP 3

#define NT_TOTAL (V_DIM / 16)   // 2000 N-tiles of 16 vocab entries
#define VGROUPS  5              // grid.y
#define WAVES_PG 4              // waves per workgroup
#define NPART    (VGROUPS * WAVES_PG)          // 20 (m,s) partials per row
#define NT_PER_WAVE (NT_TOTAL / NPART)         // 100 (even -> NB=2 tail-free)
#define CHUNKS   (D_DIM / 8)    // 16-byte bf16 chunks per row = 256
#define MROWS    32             // rows per workgroup (2 M-tiles per wave)
#define LDS_BYTES (MROWS * D_DIM * 2)          // 128 KB dynamic LDS

typedef __attribute__((ext_vector_type(16))) __bf16 v16bf;
typedef __attribute__((ext_vector_type(8)))  float  v8f;

union Frag { v16bf v; uint4 q[2]; };

__device__ __forceinline__ unsigned short f2bf(float f) {
    unsigned u = __float_as_uint(f);
    unsigned r = u + 0x7FFFu + ((u >> 16) & 1u);   // round-to-nearest-even
    return (unsigned short)(r >> 16);
}
__device__ __forceinline__ float bf2f(unsigned short h) {
    return __uint_as_float(((unsigned)h) << 16);
}

// ---------------------------------------------------------------------------
// 1) RMSNorm + fold w_norm + cast to bf16.  grid = B*(S-d), block = 256
// ---------------------------------------------------------------------------
__global__ void rmsnorm_cast(const float* __restrict__ h,
                             const float* __restrict__ wn,
                             unsigned short* __restrict__ x, int Sd) {
    int r = blockIdx.x;
    int b = r / Sd, s = r - b * Sd;
    const float* src = h + (size_t)(b * S_DIM + s) * D_DIM;
    unsigned short* dst = x + (size_t)(b * S_DIM + s) * D_DIM;

    float a = 0.f;
    for (int k = threadIdx.x; k < D_DIM; k += 256) { float v = src[k]; a += v * v; }
    __shared__ float red[256];
    red[threadIdx.x] = a; __syncthreads();
    for (int o = 128; o > 0; o >>= 1) {
        if (threadIdx.x < o) red[threadIdx.x] += red[threadIdx.x + o];
        __syncthreads();
    }
    float scale = rsqrtf(red[0] / (float)D_DIM + 1e-6f);
    for (int k = threadIdx.x; k < D_DIM; k += 256)
        dst[k] = f2bf(src[k] * scale * wn[k]);
}

// ---------------------------------------------------------------------------
// 2) Cast one w_proj slice (V x D fp32) to bf16.  float4 -> ushort4
// ---------------------------------------------------------------------------
__global__ void cast_w_bf16(const float* __restrict__ w,
                            unsigned short* __restrict__ o, long long n4) {
    long long i = (long long)blockIdx.x * blockDim.x + threadIdx.x;
    long long stride = (long long)gridDim.x * blockDim.x;
    const float4* src = (const float4*)w;
    ushort4* dst = (ushort4*)o;
    for (; i < n4; i += stride) {
        float4 f = src[i];
        ushort4 u;
        u.x = f2bf(f.x); u.y = f2bf(f.y); u.z = f2bf(f.z); u.w = f2bf(f.w);
        dst[i] = u;
    }
}

// ---------------------------------------------------------------------------
// 3) Fused GEMM + streaming logsumexp, register-blocked MB=2 x NB=2.
//    grid = (ceil(rows/32), VGROUPS), block = 128 (4 wave32), 128KB dyn LDS.
//    32xD bf16 A-tile staged once (XOR-swizzled 16B chunks -> conflict-free
//    ds_load_b128). Each B fragment feeds 2 WMMAs (halves L2 traffic), each
//    A fragment feeds 2 WMMAs (halves LDS traffic): 4 WMMAs per K-chunk.
// ---------------------------------------------------------------------------
__global__ void __launch_bounds__(128)
gemm_lse(const unsigned short* __restrict__ x,
         const unsigned short* __restrict__ wp,
         float* __restrict__ pm, float* __restrict__ ps,
         int rows, int Sd) {
    extern __shared__ unsigned short tileA[];   // 32 rows x 2048 bf16 = 128 KB

    int mtile = blockIdx.x;

    // Cooperative A-tile stage: 32 rows x 256 chunks of 16B, XOR swizzle on chunk idx
    for (int q = threadIdx.x; q < MROWS * CHUNKS; q += 128) {
        int row = q >> 8;          // /256
        int c   = q & (CHUNKS - 1);
        int g = mtile * MROWS + row; if (g >= rows) g = rows - 1;
        int b = g / Sd, s = g - b * Sd;
        const uint4* src = (const uint4*)(x + (size_t)(b * S_DIM + s) * D_DIM);
        ((uint4*)tileA)[row * CHUNKS + (c ^ (row & 15))] = src[c];
    }
    __syncthreads();

    int wave = threadIdx.x >> 5;
    int lane = threadIdx.x & 31;
    int half = lane >> 4;          // K-offset selector per ISA A/B layout
    int rA   = lane & 15;          // matrix row (A) / vocab column (B)

    int pidx = blockIdx.y * WAVES_PG + wave;       // 0..19
    int nt0  = pidx * NT_PER_WAVE;

    float m[2][8], s[2][8];
#pragma unroll
    for (int t = 0; t < 2; ++t)
#pragma unroll
        for (int i = 0; i < 8; ++i) { m[t][i] = -1e30f; s[t][i] = 0.f; }

    const uint4* ldsq = (const uint4*)tileA;

    for (int nt = 0; nt < NT_PER_WAVE; nt += 2) {
        int n0 = (nt0 + nt) * 16;
        const uint4* bq0 = (const uint4*)(wp + (size_t)(n0 + rA) * D_DIM);
        const uint4* bq1 = (const uint4*)(wp + (size_t)(n0 + 16 + rA) * D_DIM);

        v8f acc00 = {}, acc01 = {};    // M-tile 0 x {N0, N1}
        v8f acc10 = {}, acc11 = {};    // M-tile 1 x {N0, N1}
#pragma unroll 2
        for (int kc = 0; kc < D_DIM / 32; ++kc) {
            int c0 = (kc << 2) + half;          // 16B chunk index within row
            int cs0 = c0 ^ rA, cs1 = (c0 + 2) ^ rA;
            Frag a0, a1, b0, b1;
            a0.q[0] = ldsq[rA * CHUNKS + cs0];
            a0.q[1] = ldsq[rA * CHUNKS + cs1];
            a1.q[0] = ldsq[(16 + rA) * CHUNKS + cs0];
            a1.q[1] = ldsq[(16 + rA) * CHUNKS + cs1];
            b0.q[0] = bq0[c0];
            b0.q[1] = bq0[c0 + 2];
            b1.q[0] = bq1[c0];
            b1.q[1] = bq1[c0 + 2];
            acc00 = __builtin_amdgcn_wmma_f32_16x16x32_bf16(
                false, a0.v, false, b0.v, (short)0, acc00, false, false);
            acc01 = __builtin_amdgcn_wmma_f32_16x16x32_bf16(
                false, a0.v, false, b1.v, (short)0, acc01, false, false);
            acc10 = __builtin_amdgcn_wmma_f32_16x16x32_bf16(
                false, a1.v, false, b0.v, (short)0, acc10, false, false);
            acc11 = __builtin_amdgcn_wmma_f32_16x16x32_bf16(
                false, a1.v, false, b1.v, (short)0, acc11, false, false);
        }
        // online (max, sumexp) update: per-lane column subset of each row
#pragma unroll
        for (int i = 0; i < 8; ++i) {
            float v0 = acc00[i], v1 = acc01[i];
            float nm = fmaxf(m[0][i], fmaxf(v0, v1));
            s[0][i] = s[0][i] * __expf(m[0][i] - nm) + __expf(v0 - nm) + __expf(v1 - nm);
            m[0][i] = nm;
            float w0 = acc10[i], w1 = acc11[i];
            float nw = fmaxf(m[1][i], fmaxf(w0, w1));
            s[1][i] = s[1][i] * __expf(m[1][i] - nw) + __expf(w0 - nw) + __expf(w1 - nw);
            m[1][i] = nw;
        }
    }

    // Merge across the 16 lanes of each half (C layout: lanes 0-15 rows 0-7, 16-31 rows 8-15)
#pragma unroll
    for (int off = 1; off < 16; off <<= 1) {
#pragma unroll
        for (int t = 0; t < 2; ++t)
#pragma unroll
            for (int i = 0; i < 8; ++i) {
                float om = __shfl_xor(m[t][i], off);
                float os = __shfl_xor(s[t][i], off);
                float nm = fmaxf(m[t][i], om);
                s[t][i] = s[t][i] * __expf(m[t][i] - nm) + os * __expf(om - nm);
                m[t][i] = nm;
            }
    }

    if ((lane & 15) == 0) {
#pragma unroll
        for (int t = 0; t < 2; ++t)
#pragma unroll
            for (int i = 0; i < 8; ++i) {
                int row = mtile * MROWS + t * 16 + half * 8 + i;
                if (row < rows) {
                    pm[(size_t)row * NPART + pidx] = m[t][i];
                    ps[(size_t)row * NPART + pidx] = s[t][i];
                }
            }
    }
}

// ---------------------------------------------------------------------------
// 4) Per-row finalize: merge 20 (m,s) partials -> lse; target-logit dot; row value
//    grid = rows, block = 128
// ---------------------------------------------------------------------------
__global__ void finalize_rows(const float* __restrict__ pm,
                              const float* __restrict__ ps,
                              const unsigned short* __restrict__ x,
                              const unsigned short* __restrict__ wp,
                              const int* __restrict__ ids,
                              float* __restrict__ rowv, int Sd, int d) {
    int r = blockIdx.x;
    int b = r / Sd, s = r - b * Sd;
    const unsigned short* xr = x + (size_t)(b * S_DIM + s) * D_DIM;
    int tgt = ids[b * S_DIM + s + d];
    const unsigned short* wr = wp + (size_t)tgt * D_DIM;

    float a = 0.f;
    for (int k = threadIdx.x; k < D_DIM; k += 128)
        a += bf2f(xr[k]) * bf2f(wr[k]);
    __shared__ float red[128];
    red[threadIdx.x] = a; __syncthreads();
    for (int o = 64; o > 0; o >>= 1) {
        if (threadIdx.x < o) red[threadIdx.x] += red[threadIdx.x + o];
        __syncthreads();
    }
    if (threadIdx.x == 0) {
        float M = -1e30f;
        for (int p = 0; p < NPART; ++p) M = fmaxf(M, pm[(size_t)r * NPART + p]);
        float S = 0.f;
        for (int p = 0; p < NPART; ++p)
            S += ps[(size_t)r * NPART + p] * __expf(pm[(size_t)r * NPART + p] - M);
        rowv[r] = (M + __logf(S)) - red[0];
    }
}

// ---------------------------------------------------------------------------
// 5) Deterministic single-block reduction; d==1 overwrites, else accumulates
// ---------------------------------------------------------------------------
__global__ void reduce_add(const float* __restrict__ rowv, int n, float scale,
                           float* __restrict__ out, int first) {
    __shared__ float red[256];
    float a = 0.f;
    for (int i = threadIdx.x; i < n; i += 256) a += rowv[i];
    red[threadIdx.x] = a; __syncthreads();
    for (int o = 128; o > 0; o >>= 1) {
        if (threadIdx.x < o) red[threadIdx.x] += red[threadIdx.x + o];
        __syncthreads();
    }
    if (threadIdx.x == 0) {
        if (first) out[0] = scale * red[0];
        else       out[0] += scale * red[0];
    }
}

// ---------------------------------------------------------------------------
extern "C" void kernel_launch(void* const* d_in, const int* in_sizes, int n_in,
                              void* d_out, int out_size, void* d_ws, size_t ws_size,
                              hipStream_t stream) {
    const float* hidden = (const float*)d_in[0];   // (B,S,D) fp32
    const float* wnorm  = (const float*)d_in[1];   // (K,D)   fp32
    const float* wproj  = (const float*)d_in[2];   // (K,V,D) fp32
    const int*   ids    = (const int*)d_in[3];     // (B,S)   int
    float* out = (float*)d_out;

    char* ws = (char*)d_ws;
    const size_t SZ_X  = (size_t)B_DIM * S_DIM * D_DIM * 2;   // 16 MB bf16 x
    const size_t SZ_W  = (size_t)V_DIM * D_DIM * 2;           // 131 MB bf16 w slice
    const size_t SZ_PM = (size_t)4096 * NPART * 4;            // partial buffers
    unsigned short* x_bf = (unsigned short*)(ws);
    unsigned short* w_bf = (unsigned short*)(ws + SZ_X);
    float* pm   = (float*)(ws + SZ_X + SZ_W);
    float* ps   = (float*)(ws + SZ_X + SZ_W + SZ_PM);
    float* rowv = (float*)(ws + SZ_X + SZ_W + 2 * SZ_PM);

    for (int d = 1; d <= K_MTP; ++d) {
        int Sd = S_DIM - d;
        int rows = B_DIM * Sd;
        int mtiles = (rows + MROWS - 1) / MROWS;

        rmsnorm_cast<<<rows, 256, 0, stream>>>(
            hidden, wnorm + (size_t)(d - 1) * D_DIM, x_bf, Sd);

        cast_w_bf16<<<4096, 256, 0, stream>>>(
            wproj + (size_t)(d - 1) * V_DIM * D_DIM, w_bf,
            (long long)V_DIM * D_DIM / 4);

        dim3 grid(mtiles, VGROUPS);
        gemm_lse<<<grid, 128, LDS_BYTES, stream>>>(x_bf, w_bf, pm, ps, rows, Sd);

        finalize_rows<<<rows, 128, 0, stream>>>(pm, ps, x_bf, w_bf, ids, rowv, Sd, d);

        reduce_add<<<1, 256, 0, stream>>>(
            rowv, rows, 0.3f / (float)(K_MTP * rows), out, d == 1);
    }
}
